// Koopman_OPT_59639915872844
// MI455X (gfx1250) — matrix operand order repaired and checked by reading, not compiled, not run on hardware
//
#include <hip/hip_runtime.h>

typedef __attribute__((ext_vector_type(16))) _Float16 v16h;
typedef __attribute__((ext_vector_type(8)))  _Float16 v8h;
typedef __attribute__((ext_vector_type(4)))  _Float16 v4h;
typedef __attribute__((ext_vector_type(2)))  _Float16 v2h;
typedef __attribute__((ext_vector_type(8)))  float    v8f;
typedef __attribute__((ext_vector_type(4)))  float    v4f;

#define KD      128                 // koopman dim (2*NUM_EIG)
#define BATCH   131072
#define ROWS    32                  // rows per wave-job (two 16-row M-tiles)
#define NJOBS   (BATCH / ROWS)      // 4096
#define DT_C    0.01f
#define WPB     8                   // waves per block (wave32)
#define BLOCK   (WPB * 32)
#define GRID    256                 // 2048 waves -> 2 jobs/wave

#define WMMA_F16(A, B, C) \
    __builtin_amdgcn_wmma_f32_16x16x32_f16(false, (A), false, (B), (short)0, (C), false, false)

__device__ __forceinline__ float fast_tanh(float v) {
#if __has_builtin(__builtin_amdgcn_tanhf)
    return __builtin_amdgcn_tanhf(v);          // v_tanh_f32 (gfx1250 trans op)
#elif __has_builtin(__builtin_amdgcn_tanh_f32)
    return __builtin_amdgcn_tanh_f32(v);
#else
    float e = __expf(2.0f * v);
    return 1.0f - 2.0f * __builtin_amdgcn_rcpf(e + 1.0f);
#endif
}

__global__ __launch_bounds__(BLOCK) void koopman_wmma(
    const float* __restrict__ x,  const float* __restrict__ W1,
    const float* __restrict__ b1, const float* __restrict__ W2,
    const float* __restrict__ b2, float* __restrict__ y)
{
    // B-fragments of W1/W2 in WMMA layout: 2 layers * 32 frags * 512 halfs = 64 KB
    __shared__ _Float16 sW[2 * 32 * 512];
    // per-wave 32x128 f16 scratch for the inter-GEMM transpose: 64 KB total
    __shared__ _Float16 sH[WPB * ROWS * KD];

    const int tid  = threadIdx.x;
    const int lane = tid & 31;
    const int wib  = tid >> 5;
    const int lo   = lane & 15;
    const int hi   = lane >> 4;

    // ---- stage W1/W2 (f32 row-major) -> f16 WMMA B-fragments in LDS ----
    // B layout (16x16x32): lane L holds column N=L&15; lanes 0-15 cover
    // K = {kb..kb+7, kb+16..kb+23}, lanes 16-31 cover K = {kb+8..kb+15, kb+24..kb+31}.
    for (int e = tid; e < 2 * KD * KD; e += BLOCK) {
        const int layer = e >> 14;
        const int rem   = e & (KD * KD - 1);
        const int k = rem >> 7;
        const int n = rem & (KD - 1);
        const float w = layer ? W2[rem] : W1[rem];
        const int f = layer * 32 + (n >> 4) * 4 + (k >> 5);   // frag id
        const int l = ((k >> 3) & 1) * 16 + (n & 15);         // lane in frag
        sW[f * 512 + l * 16 + (k & 7) + ((k >> 4) & 1) * 8] = (_Float16)w;
    }
    __syncthreads();

    // per-lane bias registers (C-accumulator init)
    float bias1[8], bias2[8];
#pragma unroll
    for (int nt = 0; nt < 8; ++nt) {
        bias1[nt] = b1[nt * 16 + lo];
        bias2[nt] = b2[nt * 16 + lo];
    }

    _Float16* hBuf = &sH[wib * ROWS * KD];
    const int nWaves = (GRID * BLOCK) >> 5;
    const int waveId = (blockIdx.x * BLOCK + tid) >> 5;

    for (int job = waveId; job < NJOBS; job += nWaves) {
        // ---- prefetch next job's x tile (global_prefetch_b8), 128B/lane steps ----
        if (job + nWaves < NJOBS) {
            const float* nx = x + (size_t)(job + nWaves) * ROWS * KD;
#pragma unroll
            for (int g = 0; g < 4; ++g)
                __builtin_prefetch(nx + (size_t)(g * 32 + lane) * 32, 0, 0);
        }

        // ---- A-fragments of x (f32 -> f16), ISA 16-bit A layout ----
        // lanes 0-15: row lo, K = {kb..kb+7, kb+16..kb+23}
        // lanes 16-31: row lo, K = {kb+8..kb+15, kb+24..kb+31}
        v16h a1[2][4];
#pragma unroll
        for (int t = 0; t < 2; ++t) {
            const float* xrow = x + (size_t)(job * ROWS + t * 16 + lo) * KD;
#pragma unroll
            for (int kk = 0; kk < 4; ++kk) {
                const int s = kk * 32 + hi * 8;
                v4f f0 = *(const v4f*)(xrow + s);
                v4f f1 = *(const v4f*)(xrow + s + 4);
                v4f f2 = *(const v4f*)(xrow + s + 16);
                v4f f3 = *(const v4f*)(xrow + s + 20);
                v16h av;
#pragma unroll
                for (int j = 0; j < 4; ++j) {
                    av[j]      = (_Float16)f0[j];
                    av[4 + j]  = (_Float16)f1[j];
                    av[8 + j]  = (_Float16)f2[j];
                    av[12 + j] = (_Float16)f3[j];
                }
                a1[t][kk] = av;
            }
        }

        // ---- layer 1: h = tanh(x @ W1 + b1) -> hBuf (f16 row-major 32x128) ----
        // Two independent accumulator chains share each B-fragment (2-way WMMA ILP).
#pragma unroll
        for (int nt = 0; nt < 8; ++nt) {
            v8f c0, c1;
#pragma unroll
            for (int i = 0; i < 8; ++i) { c0[i] = bias1[nt]; c1[i] = bias1[nt]; }
#pragma unroll
            for (int kk = 0; kk < 4; ++kk) {
                v16h bf = *(const v16h*)&sW[(nt * 4 + kk) * 512 + lane * 16];
                c0 = WMMA_F16(a1[0][kk], bf, c0);
                c1 = WMMA_F16(a1[1][kk], bf, c1);
            }
            // Phase 1: all tanh back-to-back (independent trans ops, no hazard nops)
            float t0[8], t1[8];
#pragma unroll
            for (int i = 0; i < 8; ++i) { t0[i] = fast_tanh(c0[i]); t1[i] = fast_tanh(c1[i]); }
            // Phase 2: pack row-pairs to f16 (v_cvt_pk) and store b16 lo/hi.
            // D layout: value i is (M = i + 8*hi, N = nt*16 + lo)
            _Float16* hb = hBuf + nt * 16 + lo;
#pragma unroll
            for (int i = 0; i < 8; i += 2) {
                v2h p0, p1;
                p0[0] = (_Float16)t0[i]; p0[1] = (_Float16)t0[i + 1];
                p1[0] = (_Float16)t1[i]; p1[1] = (_Float16)t1[i + 1];
                hb[(i + 8 * hi) * KD]          = p0[0];
                hb[(i + 1 + 8 * hi) * KD]      = p0[1];
                hb[(16 + i + 8 * hi) * KD]     = p1[0];
                hb[(16 + i + 1 + 8 * hi) * KD] = p1[1];
            }
        }

        // ---- A-fragments of h from LDS (wave-local; LDS is in-order per wave) ----
        v16h a2[2][4];
#pragma unroll
        for (int t = 0; t < 2; ++t) {
#pragma unroll
            for (int kk = 0; kk < 4; ++kk) {
                const _Float16* hr = hBuf + (t * 16 + lo) * KD + kk * 32 + hi * 8;
                v8h r0 = *(const v8h*)(hr);
                v8h r1 = *(const v8h*)(hr + 16);
                a2[t][kk] = __builtin_shufflevector(r0, r1,
                                0,1,2,3,4,5,6,7,8,9,10,11,12,13,14,15);
            }
        }

        // ---- layer 2: h2 = h @ W2 + b2 -> hBuf (overwrite, f16) ----
#pragma unroll
        for (int nt = 0; nt < 8; ++nt) {
            v8f c0, c1;
#pragma unroll
            for (int i = 0; i < 8; ++i) { c0[i] = bias2[nt]; c1[i] = bias2[nt]; }
#pragma unroll
            for (int kk = 0; kk < 4; ++kk) {
                v16h bf = *(const v16h*)&sW[(32 + nt * 4 + kk) * 512 + lane * 16];
                c0 = WMMA_F16(a2[0][kk], bf, c0);
                c1 = WMMA_F16(a2[1][kk], bf, c1);
            }
            _Float16* hb = hBuf + nt * 16 + lo;
#pragma unroll
            for (int i = 0; i < 8; i += 2) {
                v2h p0, p1;
                p0[0] = (_Float16)c0[i]; p0[1] = (_Float16)c0[i + 1];
                p1[0] = (_Float16)c1[i]; p1[1] = (_Float16)c1[i + 1];
                hb[(i + 8 * hi) * KD]          = p0[0];
                hb[(i + 1 + 8 * hi) * KD]      = p0[1];
                hb[(16 + i + 8 * hi) * KD]     = p1[0];
                hb[(16 + i + 1 + 8 * hi) * KD] = p1[1];
            }
        }

        // ---- elementwise Koopman 2x2 rotation-scaling, float4-coalesced ----
        // mu = h2[:, 2i], omega = h2[:, 2i+1] are adjacent -> no shuffles.
        const float* xt = x + (size_t)job * ROWS * KD;
        float*       yt = y + (size_t)job * ROWS * KD;
#pragma unroll
        for (int g = 0; g < 32; ++g) {
            const int e4 = (g * 32 + lane) * 4;         // float idx within job tile
            v4f xv = *(const v4f*)(xt + e4);
            v4h hv = *(const v4h*)(hBuf + e4);
            v4f out;
#pragma unroll
            for (int p = 0; p < 2; ++p) {
                float mu = (float)hv[2 * p];
                float om = (float)hv[2 * p + 1];
                float ex = __expf(DT_C * mu);
                float sn, cs;
                __sincosf(DT_C * om, &sn, &cs);
                float xe = xv[2 * p], xo = xv[2 * p + 1];
                out[2 * p]     = ex * (cs * xe - sn * xo);
                out[2 * p + 1] = ex * (sn * xe + cs * xo);
            }
            *(v4f*)(yt + e4) = out;
        }
    }
}

extern "C" void kernel_launch(void* const* d_in, const int* in_sizes, int n_in,
                              void* d_out, int out_size, void* d_ws, size_t ws_size,
                              hipStream_t stream) {
    const float* x  = (const float*)d_in[0];
    const float* W1 = (const float*)d_in[1];
    const float* b1 = (const float*)d_in[2];
    const float* W2 = (const float*)d_in[3];
    const float* b2 = (const float*)d_in[4];
    float* y = (float*)d_out;
    koopman_wmma<<<GRID, BLOCK, 0, stream>>>(x, W1, b1, W2, b2, y);
}
